// ReinforceSampler_35828617183722
// MI455X (gfx1250) — matrix operand order repaired and checked by reading, not compiled, not run on hardware
//
#include <hip/hip_runtime.h>
#include <stdint.h>

typedef __attribute__((ext_vector_type(16))) _Float16 v16h;
typedef __attribute__((ext_vector_type(8)))  float    v8f;
typedef __attribute__((ext_vector_type(4)))  unsigned int v4u;
typedef __attribute__((ext_vector_type(4)))  int      v4i;
typedef __attribute__((ext_vector_type(8)))  int      v8i;

#define SDIM 256
#define BDIM 32
#define AITER 4
#define DDIM 512
#define NITER 8

#if defined(__has_builtin)
#if __has_builtin(__builtin_amdgcn_tensor_load_to_lds) && __has_builtin(__builtin_amdgcn_s_wait_tensorcnt)
#define HAVE_TDM 1
#endif
#endif

// ---------------------------------------------------------------------------
// M = I + L, and Gershgorin bound: maxbits[b] = max_r sum_c |M[r,c]| (as bits)
// ---------------------------------------------------------------------------
__global__ __launch_bounds__(256) void k_build_M(const float* __restrict__ L,
                                                 float* __restrict__ M,
                                                 unsigned int* __restrict__ maxbits) {
  int b = blockIdx.x / SDIM, r = blockIdx.x % SDIM, c = threadIdx.x;
  size_t base = (size_t)b * SDIM * SDIM;
  float v = L[base + r * SDIM + c] + (r == c ? 1.0f : 0.0f);
  M[base + r * SDIM + c] = v;
  __shared__ float red[256];
  red[c] = fabsf(v);
  __syncthreads();
  for (int off = 128; off > 0; off >>= 1) {
    if (c < off) red[c] += red[c + off];
    __syncthreads();
  }
  if (c == 0) atomicMax(&maxbits[b], __float_as_uint(red[0]));
}

// X0 = alpha * I,  alpha = 1 / rowSumBound  (spectral radius of I - alpha*M < 1)
__global__ __launch_bounds__(256) void k_init_X(float* __restrict__ X,
                                                const unsigned int* __restrict__ maxbits) {
  int b = blockIdx.x / SDIM, r = blockIdx.x % SDIM, c = threadIdx.x;
  float alpha = 1.0f / __uint_as_float(maxbits[b]);
  X[(size_t)b * SDIM * SDIM + r * SDIM + c] = (r == c) ? alpha : 0.0f;
}

// ---------------------------------------------------------------------------
// LDS-staged WMMA GEMM, 64x64 block tile, 8 wave32s (2x4), 32x16 per wave.
//   MODE 0: C = A*B              MODE 1: C = 2*E - A*B  (Newton-Schulz)
// LDS panels are stored in WMMA fragment order so each lane reads its whole
// 16-half fragment as one contiguous 32-byte LDS access.
//   A-frag (16x32 f16): lane = (row&15) + 16*g, K in {g*8..+7} u {16+g*8..+7}
//   B-frag (32x16 f16): lane = (col&15) + 16*g, K in {g*8..+7} u {16+g*8..+7}
//   C/D   (16x16 f32): row = v + 8*(lane>>4), col = lane&15
// ---------------------------------------------------------------------------
template <int MODE>
__global__ __launch_bounds__(256) void k_gemm(const float* __restrict__ Am,
                                              const float* __restrict__ Bm,
                                              float* __restrict__ Cm,
                                              const float* __restrict__ Em) {
  __shared__ __align__(32) _Float16 Apan[4 * 32 * 16];  // [rowTile][lane][half]
  __shared__ __align__(32) _Float16 Bpan[4 * 32 * 16];  // [colTile][lane][half]

  int batch = blockIdx.x >> 4;
  int btile = blockIdx.x & 15;
  int brow = (btile >> 2) * 64;
  int bcol = (btile & 3) * 64;
  int tid = threadIdx.x;
  int wave = tid >> 5, lane = tid & 31;
  int wrow = wave >> 2, wcol = wave & 3;  // 2x4 wave grid
  int g = lane >> 4, q = lane & 15;

  const float* Ab = Am + (size_t)batch * SDIM * SDIM;
  const float* Bb = Bm + (size_t)batch * SDIM * SDIM;

  v8f acc0 = {0.f, 0.f, 0.f, 0.f, 0.f, 0.f, 0.f, 0.f};
  v8f acc1 = acc0;

  for (int k0 = 0; k0 < SDIM; k0 += 32) {
    // ---- stage A panel (64 x 32) and B panel (32 x 64), f32 -> f16 --------
#pragma unroll
    for (int rep = 0; rep < 2; ++rep) {
      int f = tid + rep * 256;            // 512 float4 per panel
      {                                    // A: row = f>>3, kq = (f&7)*4
        int r = f >> 3, kq = (f & 7) * 4;
        float4 vv = *(const float4*)&Ab[(size_t)(brow + r) * SDIM + k0 + kq];
        int rt = r >> 4, rq = r & 15;
        int ag = ((kq & 15) >= 8) ? 1 : 0;
        int al = rq + 16 * ag;
        int h0 = (kq < 16) ? (kq & 7) : (8 + (kq & 7));
        _Float16* dstp = &Apan[(rt * 32 + al) * 16 + h0];
        dstp[0] = (_Float16)vv.x; dstp[1] = (_Float16)vv.y;
        dstp[2] = (_Float16)vv.z; dstp[3] = (_Float16)vv.w;
      }
      {                                    // B: k = f>>4, cq = (f&15)*4
        int k = f >> 4, cq = (f & 15) * 4;
        float4 vv = *(const float4*)&Bb[(size_t)(k0 + k) * SDIM + bcol + cq];
        int bg = (k >> 3) & 1, h = (k & 7) + ((k >> 4) << 3);
        float e[4] = {vv.x, vv.y, vv.z, vv.w};
#pragma unroll
        for (int i = 0; i < 4; ++i) {
          int cl = cq + i, u = cl >> 4, cqq = cl & 15;
          Bpan[(u * 32 + (cqq + 16 * bg)) * 16 + h] = (_Float16)e[i];
        }
      }
    }
    if (k0 + 32 < SDIM) {  // gfx1250 global_prefetch_b8 for next K panel
      __builtin_prefetch(&Ab[(size_t)(brow + (tid >> 3)) * SDIM + k0 + 32], 0, 1);
      __builtin_prefetch(&Bb[(size_t)(k0 + 32 + (tid >> 4)) * SDIM + bcol], 0, 1);
    }
    __syncthreads();

    // ---- two 16x16x32 WMMAs per wave --------------------------------------
    v16h bf = *(const v16h*)&Bpan[(wcol * 32 + lane) * 16];
    v16h a0 = *(const v16h*)&Apan[((wrow * 2 + 0) * 32 + lane) * 16];
    v16h a1 = *(const v16h*)&Apan[((wrow * 2 + 1) * 32 + lane) * 16];
    acc0 = __builtin_amdgcn_wmma_f32_16x16x32_f16(false, a0, false, bf,
                                                  (short)0, acc0, false, false);
    acc1 = __builtin_amdgcn_wmma_f32_16x16x32_f16(false, a1, false, bf,
                                                  (short)0, acc1, false, false);
    __syncthreads();
  }

  float* Cb = Cm + (size_t)batch * SDIM * SDIM;
  const float* Eb = Em + (size_t)batch * SDIM * SDIM;
  int col = bcol + wcol * 16 + q;
#pragma unroll
  for (int v = 0; v < 8; ++v) {
    int r0 = brow + wrow * 32 + v + 8 * g;
    int r1 = r0 + 16;
    float o0 = acc0[v], o1 = acc1[v];
    if (MODE == 1) {
      o0 = 2.0f * Eb[r0 * SDIM + col] - o0;
      o1 = 2.0f * Eb[r1 * SDIM + col] - o1;
    }
    Cb[r0 * SDIM + col] = o0;
    Cb[r1 * SDIM + col] = o1;
  }
}

// K = I - X  (X ~ (I+L)^-1), exp_sizes[b] += ||K||_F^2 partials
__global__ __launch_bounds__(256) void k_finalize(const float* __restrict__ X,
                                                  float* __restrict__ K,
                                                  float* __restrict__ exps) {
  int b = blockIdx.x / SDIM, r = blockIdx.x % SDIM, c = threadIdx.x;
  size_t base = (size_t)b * SDIM * SDIM;
  float v = (r == c ? 1.0f : 0.0f) - X[base + r * SDIM + c];
  K[base + r * SDIM + c] = v;
  __shared__ float red[256];
  red[c] = v * v;
  __syncthreads();
  for (int off = 128; off > 0; off >>= 1) {
    if (c < off) red[c] += red[c + off];
    __syncthreads();
  }
  if (c == 0) atomicAdd(&exps[b], red[0]);
}

// ---------------------------------------------------------------------------
// Sequential DPP sampler: one block per (b, alpha) sample; full 256x256 K in
// LDS (256KB < 320KB/WGP). The K tile is fetched by the Tensor Data Mover
// (tensor_load_to_lds + s_wait_tensorcnt) when the builtin exists.
// ---------------------------------------------------------------------------
__device__ __forceinline__ unsigned int fmix32(unsigned int h) {
  h ^= h >> 16; h *= 0x85ebca6bu; h ^= h >> 13; h *= 0xc2b2ae35u; h ^= h >> 16;
  return h;
}

__global__ __launch_bounds__(256) void k_dpp(const float* __restrict__ Kg,
                                             const int* __restrict__ seedp,
                                             float* __restrict__ subs) {
  extern __shared__ float sm[];
  float* Kc = sm;                 // 65536 floats (256KB)
  float* dg = sm + 65536;         // 256  (initial diagonal)
  float* zs = dg + 256;           // 256  (subset indicators)
  float* mm = zs + 256;           // misc broadcast slots

  int tid = threadIdx.x;
  int sidx = blockIdx.x;          // b*AITER + a
  int batch = sidx >> 2;
  const float* src = Kg + (size_t)batch * SDIM * SDIM;

#if defined(HAVE_TDM)
  if (tid < 32) {  // one wave issues the DMA; EXEC-independent, SGPR D#
    unsigned long long ga = (unsigned long long)(uintptr_t)src;
    unsigned int lds_off = (unsigned int)(uintptr_t)Kc;
    // D# group 0: count=1 | lds_addr | global_addr | type=2
    v4u g0;
    g0[0] = 1u;                                   // count=1, user mode
    g0[1] = lds_off;                              // lds_addr [63:32]
    g0[2] = (unsigned int)(ga & 0xffffffffu);     // global_addr [95:64]
    g0[3] = (unsigned int)((ga >> 32) & 0x01ffffffu) | (2u << 30);  // type=2
    // D# group 1: data_size=4B, 256x256 tile of a 256-stride tensor
    v8i g1;
    g1[0] = (int)(2u << 16);        // workgroup_mask=0, data_size=2 (4B)
    g1[1] = (int)(256u << 16);      // tensor_dim0[15:0] << 16   (=256)
    g1[2] = (int)(256u << 16);      // tensor_dim0 hi=0 | tensor_dim1 lo (=256)
    g1[3] = (int)(256u << 16);      // tensor_dim1 hi=0 | tile_dim0 (=256)
    g1[4] = (int)256;               // tile_dim1 = 256, tile_dim2 = 0
    g1[5] = (int)256;               // tensor_dim0_stride = 256
    g1[6] = 0;                      // stride hi, tensor_dim1_stride lo
    g1[7] = 0;
    v4i gz = {0, 0, 0, 0};
#if __clang_major__ >= 23
    v8i gz8 = {0, 0, 0, 0, 0, 0, 0, 0};
    __builtin_amdgcn_tensor_load_to_lds(g0, g1, gz, gz, gz8, 0);
#else
    __builtin_amdgcn_tensor_load_to_lds(g0, g1, gz, gz, 0);
#endif
    __builtin_amdgcn_s_wait_tensorcnt(0);
  }
  __syncthreads();
#else
  {
    const float4* s4 = (const float4*)src;
    float4* d4 = (float4*)Kc;
    for (int i = tid; i < SDIM * SDIM / 4; i += 256) d4[i] = s4[i];
    __syncthreads();
  }
#endif

  dg[tid] = Kc[tid * (SDIM + 1)];
  zs[tid] = 0.0f;
  if (tid == 0) mm[1] = 0.0f;
  unsigned int seed = (unsigned int)seedp[0];

  for (int j = 0; j < SDIM; ++j) {
    __syncthreads();
    if (tid == 0) {
      float d = Kc[j * (SDIM + 1)];
      unsigned int x = seed * 0x9E3779B9u ^ ((unsigned)sidx * 0x85EBCA6Bu)
                       ^ ((unsigned)j * 0xC2B2AE35u);
      float u = (float)(fmix32(x) >> 8) * (1.0f / 16777216.0f);
      float p = fminf(fmaxf(d, 0.0f), 1.0f);
      float z = (u < p) ? 1.0f : 0.0f;
      float dz = d - (1.0f - z);
      if (fabsf(dz) < 1e-6f) dz = (dz < 0.0f) ? -1e-6f : 1e-6f;
      zs[j] = z;
      mm[0] = 1.0f / dz;
    }
    __syncthreads();
    float inv = mm[0];
    int i = j + 1 + tid;              // <=255 trailing rows, one per thread
    if (i < SDIM) {
      float kij = Kc[i * SDIM + j] * inv;
      const float* rowj = Kc + j * SDIM;
      float* rowi = Kc + i * SDIM;
#pragma unroll 4
      for (int c = j + 1; c < SDIM; ++c) rowi[c] -= kij * rowj[c];
    }
  }
  __syncthreads();
  if (zs[tid] != 0.0f) mm[1] = 1.0f;  // benign race, all write 1
  __syncthreads();
  if (tid == 0 && mm[1] == 0.0f) {    // empty subset -> argmax of initial diag
    int am = 0; float bv = dg[0];
    for (int s = 1; s < SDIM; ++s) if (dg[s] > bv) { bv = dg[s]; am = s; }
    zs[am] = 1.0f;
  }
  __syncthreads();
  subs[(size_t)sidx * SDIM + tid] = zs[tid];
}

// output[b,a,s,:] = subs[b,a,s] * words[b,s,:]   (float4, bandwidth-bound)
__global__ __launch_bounds__(256) void k_write(const float* __restrict__ words,
                                               const float* __restrict__ subs,
                                               float* __restrict__ out) {
  size_t i = (size_t)blockIdx.x * 256 + threadIdx.x;  // float4 index
  int d4 = (int)(i & 127);
  int s  = (int)((i >> 7) & 255);
  int ba = (int)(i >> 15);                            // 0..127
  int b  = ba >> 2;
  float z = subs[(size_t)ba * SDIM + s];
  float4 wv = ((const float4*)words)[((size_t)(b * SDIM + s)) * 128 + d4];
  float4 o; o.x = wv.x * z; o.y = wv.y * z; o.z = wv.z * z; o.w = wv.w * z;
  ((float4*)out)[i] = o;
}

// ---------------------------------------------------------------------------
extern "C" void kernel_launch(void* const* d_in, const int* in_sizes, int n_in,
                              void* d_out, int out_size, void* d_ws, size_t ws_size,
                              hipStream_t stream) {
  (void)in_sizes; (void)n_in; (void)out_size; (void)ws_size;
  const float* Lk    = (const float*)d_in[0];   // [32,256,256] f32 (L-ensemble)
  const float* words = (const float*)d_in[1];   // [32,256,512] f32
  const int*   seedp = (const int*)d_in[2];     // scalar seed on device
  float* out = (float*)d_out;

  const size_t matElems = (size_t)BDIM * SDIM * SDIM;   // 2,097,152
  float* M  = (float*)d_ws;
  float* X  = M  + matElems;
  float* Y  = X  + matElems;
  float* X2 = Y  + matElems;
  unsigned int* maxbits = (unsigned int*)(X2 + matElems);

  const size_t outElems  = (size_t)BDIM * AITER * SDIM * DDIM;  // 16,777,216
  float* subs = out + outElems;                                  // [B,A,S]
  float* exps = subs + (size_t)BDIM * AITER * SDIM;              // [B]

  hipMemsetAsync(maxbits, 0, BDIM * sizeof(unsigned int), stream);
  hipMemsetAsync(exps, 0, BDIM * sizeof(float), stream);

  k_build_M<<<BDIM * SDIM, 256, 0, stream>>>(Lk, M, maxbits);
  k_init_X <<<BDIM * SDIM, 256, 0, stream>>>(X, maxbits);

  // Newton-Schulz: X <- X(2I - M X); quadratic convergence, 8 iters
  float* Xc = X; float* Xo = X2;
  for (int it = 0; it < NITER; ++it) {
    k_gemm<0><<<BDIM * 16, 256, 0, stream>>>(M,  Xc, Y,  Xc);  // Y = M*X
    k_gemm<1><<<BDIM * 16, 256, 0, stream>>>(Xc, Y,  Xo, Xc);  // Xo = 2X - X*Y
    float* t = Xc; Xc = Xo; Xo = t;
  }

  float* Kbuf = Y;  // reuse Y as the marginal kernel K = I - (I+L)^-1
  k_finalize<<<BDIM * SDIM, 256, 0, stream>>>(Xc, Kbuf, exps);

  size_t shbytes = (size_t)(65536 + 256 + 256 + 16) * sizeof(float);  // ~258KB LDS
  (void)hipFuncSetAttribute(reinterpret_cast<const void*>(k_dpp),
                            hipFuncAttributeMaxDynamicSharedMemorySize,
                            (int)shbytes);
  k_dpp<<<BDIM * AITER, 256, shbytes, stream>>>(Kbuf, seedp, subs);

  k_write<<<(int)(outElems / 4 / 256), 256, 0, stream>>>(words, subs, out);
}
